// InternLM2Attention_36979668418950
// MI455X (gfx1250) — matrix operand order repaired and checked by reading, not compiled, not run on hardware
//
#include <hip/hip_runtime.h>

// ---------------------------------------------------------------------------
// InternLM2 attention block for gfx1250 (MI455X): bf16 WMMA + TDM staging.
//   B=2, S=2048, HID=6144, H=48, KVH=8, G=6, D=128, rope theta = 1e6
// ---------------------------------------------------------------------------

#define BATCH   2
#define SLEN    2048
#define HID     6144
#define NH      48
#define NKVH    8
#define GQ      6            // NH / NKVH
#define DH      128
#define QKV_O   8192         // NKVH*(G+2)*DH
#define MROWS   (BATCH * SLEN)   // 4096

typedef __attribute__((ext_vector_type(16))) __bf16 v16bf;
typedef __attribute__((ext_vector_type(8)))  float  v8f;
typedef __attribute__((ext_vector_type(4)))  unsigned u32x4;
typedef __attribute__((ext_vector_type(8)))  int      i32x8;
typedef __attribute__((ext_vector_type(4)))  int      i32x4;

// ---------------- WMMA fragment loaders (ISA 7.12.2 layouts) ---------------

// A-matrix 16x32 bf16: lane&15 = row M, lane>=16 selects K pattern half.
// VGPR j holds K = 2j + (j>=4 ? 8 : 0) + 8*(lane>>4), K+1.
__device__ __forceinline__ v16bf load_a_frag(const __bf16* base, long ld) {
  const int lane = threadIdx.x & 31;
  const int m = lane & 15, hi = lane >> 4;
  union { v16bf v; unsigned u[8]; } r;
#pragma unroll
  for (int j = 0; j < 8; ++j) {
    const int kk = 2 * j + ((j >= 4) ? 8 : 0) + 8 * hi;
    r.u[j] = *(const unsigned*)(base + (long)m * ld + kk);
  }
  return r.v;
}

// B-matrix 32x16 bf16 built from a row-major [N][K] source (i.e. B = srcT):
// lane&15 = column N (source row), VGPR j holds K = 2j + (j>=4 ? 8 : 0) + 8*(lane>>4), +1.
__device__ __forceinline__ v16bf load_bT_frag(const __bf16* base, long ld) {
  const int lane = threadIdx.x & 31;
  const int n = lane & 15, hi = lane >> 4;
  union { v16bf v; unsigned u[8]; } r;
#pragma unroll
  for (int j = 0; j < 8; ++j) {
    const int kk = 2 * j + ((j >= 4) ? 8 : 0) + 8 * hi;
    r.u[j] = *(const unsigned*)(base + (long)n * ld + kk);
  }
  return r.v;
}

// K tile as B-matrix for scores (contraction = head dim). Row-clamped.
__device__ __forceinline__ v16bf load_k_frag(const __bf16* kbase, int jj, int d0) {
  const int lane = threadIdx.x & 31;
  int key = jj + (lane & 15);
  if (key > SLEN - 1) key = SLEN - 1;
  const int hi = lane >> 4;
  union { v16bf v; unsigned u[8]; } r;
#pragma unroll
  for (int j = 0; j < 8; ++j) {
    const int kk = d0 + 2 * j + ((j >= 4) ? 8 : 0) + 8 * hi;
    r.u[j] = *(const unsigned*)(kbase + (long)key * (NKVH * DH) + kk);
  }
  return r.v;
}

// V tile as B-matrix for P*V (contraction = key index). V is stored
// transposed (D x S) so consecutive keys are contiguous dwords. Clamped.
__device__ __forceinline__ v16bf load_v_frag(const __bf16* vbase, int s0, int d0) {
  const int lane = threadIdx.x & 31;
  const int d = d0 + (lane & 15);
  const int hi = lane >> 4;
  union { v16bf v; unsigned u[8]; } r;
#pragma unroll
  for (int j = 0; j < 8; ++j) {
    int s = s0 + 2 * j + ((j >= 4) ? 8 : 0) + 8 * hi;
    if (s > SLEN - 2) s = SLEN - 2;
    r.u[j] = *(const unsigned*)(vbase + (long)d * SLEN + s);
  }
  return r.v;
}

// -------------------- Tensor Data Mover: 2D tile -> LDS ---------------------
// D# per ISA 8.3/8.4: group0 = {count/lds_addr/global_addr/type},
// group1 = {data_size, tensor dims, tile dims, dim0 stride}. 2D only, so
// groups 2/3 (and the trailing group) are zero. 6-arg builtin form
// (amdgpu-toolchain / clang-23 signature).
__device__ __forceinline__ void tdm_load_tile_2d(const __bf16* gsrc, unsigned lds_off,
                                                 unsigned tile_d0, unsigned tile_d1,
                                                 unsigned long tensor_d0,
                                                 unsigned long tensor_d1,
                                                 unsigned long stride0) {
  const unsigned long ga = (unsigned long)gsrc;
  u32x4 g0;
  g0[0] = 1u;                                   // count=1 valid descriptor
  g0[1] = lds_off;                              // lds_addr (bytes)
  g0[2] = (unsigned)(ga & 0xFFFFFFFFu);         // global_addr[31:0]
  g0[3] = (unsigned)((ga >> 32) & 0x1FFFFFFu)   // global_addr[56:32]
        | (2u << 30);                           // type = 2 ("image")
  unsigned w[8];
  w[0] = (1u << 16);                            // data_size = 1 -> 2 bytes
  w[1] = (unsigned)((tensor_d0 & 0xFFFFu) << 16);          // tensor_dim0[15:0]
  w[2] = (unsigned)((tensor_d0 >> 16) & 0xFFFFu)           // tensor_dim0[31:16]
       | (unsigned)((tensor_d1 & 0xFFFFu) << 16);          // tensor_dim1[15:0]
  w[3] = (unsigned)((tensor_d1 >> 16) & 0xFFFFu)           // tensor_dim1[31:16]
       | (tile_d0 << 16);                                  // tile_dim0
  w[4] = tile_d1;                                          // tile_dim1 (dim2=0)
  w[5] = (unsigned)(stride0 & 0xFFFFFFFFu);                // dim0_stride[31:0]
  w[6] = (unsigned)((stride0 >> 32) & 0xFFFFu);            // dim0_stride[47:32]
  w[7] = 0u;
  i32x8 g1;
#pragma unroll
  for (int i = 0; i < 8; ++i) g1[i] = (int)w[i];
  i32x4 z4 = {};
  i32x8 z8 = {};
  __builtin_amdgcn_tensor_load_to_lds(g0, g1, z4, z4, z8, 0);
}

__device__ __forceinline__ unsigned lds_offset_u32(const void* p) {
  return (unsigned)(unsigned long)(__attribute__((address_space(3))) const void*)p;
}

// ---------------------------- elementwise kernels ---------------------------

__global__ __launch_bounds__(256) void cvt_f32_bf16(const float* __restrict__ src,
                                                    __bf16* __restrict__ dst, long n) {
  long i = (long)blockIdx.x * blockDim.x + threadIdx.x;
  if (i < n) dst[i] = (__bf16)src[i];
}

// qkv (f32, [B,S,KVH,G+2,D]) -> rope(q)->qb [B,S,H,D], rope(k)->kb [B,S,KVH,D],
// v -> vtb transposed [B,KVH,D,S]
__global__ __launch_bounds__(256) void rope_split_kernel(const float* __restrict__ qkv,
                                                         __bf16* __restrict__ qb,
                                                         __bf16* __restrict__ kb,
                                                         __bf16* __restrict__ vtb) {
  const long idx = (long)blockIdx.x * blockDim.x + threadIdx.x; // 2^25 threads
  const int d    = (int)(idx & 127);
  const int slot = (int)((idx >> 7) & 7);
  const int kvh  = (int)((idx >> 10) & 7);
  const int s    = (int)((idx >> 13) & 2047);
  const int b    = (int)(idx >> 24);
  const float val = qkv[idx];
  if (slot == GQ + 1) { // V: transposed store
    vtb[((long)(b * NKVH + kvh) * DH + d) * SLEN + s] = (__bf16)val;
    return;
  }
  // RoPE: out = v*cos + rotate_half(v)*sin, inv_freq = theta^(-2i/D)
  const float partner = qkv[idx + ((d < 64) ? 64 : -64)];
  const float rot = (d < 64) ? -partner : partner;
  const int fi = d & 63;
  const float invf = __expf(-0.2158673524682f * (float)fi); // ln(1e6)*2/128
  const float ang = (float)s * invf;
  const float cv = cosf(ang), sv = sinf(ang);
  const float out = val * cv + rot * sv;
  if (slot == GQ) {
    kb[((long)(b * SLEN + s) * NKVH + kvh) * DH + d] = (__bf16)out;
  } else {
    qb[((long)(b * SLEN + s) * NH + (kvh * GQ + slot)) * DH + d] = (__bf16)out;
  }
}

// ------------------------------ bf16 WMMA GEMM ------------------------------
// C[M][N] (f32) = A[M][K] (bf16 row-major) * B[N][K]^T (bf16 weight layout).
// Block: 8 waves, tile 32(M) x 512(N). The shared 32x32 A tile is DMA'd into
// a double-buffered LDS slot by the Tensor Data Mover (one wave issues,
// s_wait_tensorcnt + barrier publishes it), overlapping with the WMMA chain.
// Each wave: 2 A-fragments x 4 B-fragments -> 8 WMMAs per K-step of 32.
__global__ __launch_bounds__(256) void gemm_bf16_nt(const __bf16* __restrict__ A,
                                                    const __bf16* __restrict__ B,
                                                    float* __restrict__ C,
                                                    int M, int N, int K) {
  __shared__ __bf16 Asmem[2][32 * 32]; // double-buffered 32x32 bf16 A tile
  const int wave = threadIdx.x >> 5;
  const int lane = threadIdx.x & 31;
  const int m0 = blockIdx.y * 32;
  const int n0 = blockIdx.x * 512 + wave * 64;
  const int n = lane & 15, hi = lane >> 4;
  const __bf16* Abase = A + (long)m0 * K;

  // prologue: DMA first A tile
  if (wave == 0) {
    tdm_load_tile_2d(Abase, lds_offset_u32(&Asmem[0][0]), 32, 32,
                     (unsigned long)K, (unsigned long)(M - m0), (unsigned long)K);
    __builtin_amdgcn_s_wait_tensorcnt(0);
  }
  __syncthreads();

  v8f acc0[4] = {}, acc1[4] = {};
  int p = 0;
  for (int k0 = 0; k0 < K; k0 += 32) {
    // prefetch next A tile into the other LDS buffer while we compute
    if (wave == 0 && k0 + 32 < K)
      tdm_load_tile_2d(Abase + k0 + 32, lds_offset_u32(&Asmem[p ^ 1][0]), 32, 32,
                       (unsigned long)(K - k0 - 32), (unsigned long)(M - m0),
                       (unsigned long)K);
    __builtin_prefetch(B + (long)(n0 + n) * K + k0 + 256, 0, 3);

    // A fragments (rows 0-15 and 16-31) from LDS
    v16bf a0, a1;
#pragma unroll
    for (int j = 0; j < 8; ++j) {
      const int kk = 2 * j + ((j >= 4) ? 8 : 0) + 8 * hi;
      a0[2 * j]     = Asmem[p][n * 32 + kk];
      a0[2 * j + 1] = Asmem[p][n * 32 + kk + 1];
      a1[2 * j]     = Asmem[p][(n + 16) * 32 + kk];
      a1[2 * j + 1] = Asmem[p][(n + 16) * 32 + kk + 1];
    }
    // all four B fragments first, then the WMMA chain (one wait, 8 WMMAs)
    v16bf bf[4];
#pragma unroll
    for (int t = 0; t < 4; ++t)
      bf[t] = load_bT_frag(B + (long)(n0 + t * 16) * K + k0, K);
#pragma unroll
    for (int t = 0; t < 4; ++t)
      acc0[t] = __builtin_amdgcn_wmma_f32_16x16x32_bf16(
          false, a0, false, bf[t], (short)0, acc0[t], false, false);
#pragma unroll
    for (int t = 0; t < 4; ++t)
      acc1[t] = __builtin_amdgcn_wmma_f32_16x16x32_bf16(
          false, a1, false, bf[t], (short)0, acc1[t], false, false);

    if (wave == 0) __builtin_amdgcn_s_wait_tensorcnt(0); // next buffer landed
    __syncthreads();                                     // publish to all waves
    p ^= 1;
  }

  const int mrow = hi * 8;
#pragma unroll
  for (int t = 0; t < 4; ++t)
#pragma unroll
    for (int r = 0; r < 8; ++r) {
      C[(long)(m0 + mrow + r) * N + n0 + t * 16 + n] = acc0[t][r];
      C[(long)(m0 + 16 + mrow + r) * N + n0 + t * 16 + n] = acc1[t][r];
    }
}

// ---------------------- flash attention (one Q tile / wave) -----------------

__global__ __launch_bounds__(128) void attn_kernel(const __bf16* __restrict__ q,
                                                   const __bf16* __restrict__ k,
                                                   const __bf16* __restrict__ vt,
                                                   __bf16* __restrict__ ctx) {
  __shared__ __bf16 Plds[4][16 * 32]; // per-wave P staging tile (bf16)
  const int wave = threadIdx.x >> 5;
  const int lane = threadIdx.x & 31;
  const int gid = blockIdx.x * 4 + wave;      // B*NH*(S/16) = 12288 waves
  const int qt = gid & 127;                   // S/16 = 128 q-tiles
  const int h = (gid >> 7) % NH;
  const int b = gid / (128 * NH);
  const int kvh = h / GQ;
  const int n = lane & 15, hi = lane >> 4;

  const __bf16* qbase = q + ((long)(b * SLEN + qt * 16) * NH + h) * DH;
  const __bf16* kbase = k + (long)b * SLEN * NKVH * DH + (long)kvh * DH;
  const __bf16* vbase = vt + ((long)(b * NKVH + kvh) * DH) * SLEN;

  // Q tile (16x128) as 4 A-fragments, kept in registers for the whole pass.
  v16bf qa[4];
#pragma unroll
  for (int kk = 0; kk < 4; ++kk) qa[kk] = load_a_frag(qbase + kk * 32, NH * DH);

  v8f acc[8] = {};
  float mrun[8], lrun[8];
#pragma unroll
  for (int r = 0; r < 8; ++r) { mrun[r] = -3.0e38f; lrun[r] = 0.0f; }
  const float scale = 0.08838834764831845f; // 1/sqrt(128)
  const int qrow0 = qt * 16;

  for (int j0 = 0; j0 <= qrow0; j0 += 32) {
    // ---- scores: two 16-key subtiles, 4 WMMAs each over D=128 ----
    v8f s0 = {}, s1 = {};
#pragma unroll
    for (int kk = 0; kk < 4; ++kk) {
      const v16bf kf0 = load_k_frag(kbase, j0, kk * 32);
      const v16bf kf1 = load_k_frag(kbase, j0 + 16, kk * 32);
      s0 = __builtin_amdgcn_wmma_f32_16x16x32_bf16(
          false, qa[kk], false, kf0, (short)0, s0, false, false);
      s1 = __builtin_amdgcn_wmma_f32_16x16x32_bf16(
          false, qa[kk], false, kf1, (short)0, s1, false, false);
    }
    // ---- causal mask + online softmax (row = r + 8*hi across 16 lanes) ----
#pragma unroll
    for (int r = 0; r < 8; ++r) {
      const int row = qrow0 + r + 8 * hi;
      float v0 = (j0 + n <= row) ? s0[r] * scale : -1.0e30f;
      float v1 = (j0 + 16 + n <= row) ? s1[r] * scale : -1.0e30f;
      float bm = fmaxf(v0, v1);
      bm = fmaxf(bm, __shfl_xor(bm, 1, 16));
      bm = fmaxf(bm, __shfl_xor(bm, 2, 16));
      bm = fmaxf(bm, __shfl_xor(bm, 4, 16));
      bm = fmaxf(bm, __shfl_xor(bm, 8, 16));
      const float mnew = fmaxf(mrun[r], bm);
      const float sfac = __expf(mrun[r] - mnew);
      mrun[r] = mnew;
      const float p0 = __expf(v0 - mnew);
      const float p1 = __expf(v1 - mnew);
      float ps = p0 + p1;
      ps += __shfl_xor(ps, 1, 16);
      ps += __shfl_xor(ps, 2, 16);
      ps += __shfl_xor(ps, 4, 16);
      ps += __shfl_xor(ps, 8, 16);
      lrun[r] = lrun[r] * sfac + ps;
#pragma unroll
      for (int t = 0; t < 8; ++t) acc[t][r] = acc[t][r] * sfac;
      const int m = r + 8 * hi;
      Plds[wave][m * 32 + n] = (__bf16)p0;
      Plds[wave][m * 32 + 16 + n] = (__bf16)p1;
    }
    // wave-private LDS transpose: ensure ds stores land before re-reading
    asm volatile("s_wait_dscnt 0x0" ::: "memory");
    // P (16x32) back as an A-fragment
    v16bf pa;
#pragma unroll
    for (int j = 0; j < 8; ++j) {
      const int kk = 2 * j + ((j >= 4) ? 8 : 0) + 8 * hi;
      pa[2 * j] = Plds[wave][n * 32 + kk];
      pa[2 * j + 1] = Plds[wave][n * 32 + kk + 1];
    }
    // ---- ctx += P * V : 8 WMMAs across D=128 ----
#pragma unroll
    for (int t = 0; t < 8; ++t)
      acc[t] = __builtin_amdgcn_wmma_f32_16x16x32_bf16(
          false, pa, false, load_v_frag(vbase, j0, t * 16), (short)0, acc[t], false, false);
  }

  // ---- normalize + store ctx as bf16, layout (B,S,H*D) = (B,S,HID) ----
#pragma unroll
  for (int r = 0; r < 8; ++r) {
    const float inv = 1.0f / lrun[r];
    const int srow = qrow0 + r + 8 * hi;
    const long base = ((long)(b * SLEN + srow) * NH + h) * DH + n;
#pragma unroll
    for (int t = 0; t < 8; ++t)
      ctx[base + t * 16] = (__bf16)(acc[t][r] * inv);
  }
}

// ------------------------------- launcher -----------------------------------

extern "C" void kernel_launch(void* const* d_in, const int* in_sizes, int n_in,
                              void* d_out, int out_size, void* d_ws, size_t ws_size,
                              hipStream_t stream) {
  (void)in_sizes; (void)n_in; (void)out_size; (void)ws_size;
  const float* x    = (const float*)d_in[0];
  const float* Wqkv = (const float*)d_in[1];
  const float* Wo   = (const float*)d_in[2];
  float* out = (float*)d_out;

  char* ws = (char*)d_ws;
  size_t off = 0;
  auto carve = [&](size_t bytes) {
    void* p = ws + off;
    off += (bytes + 255) & ~(size_t)255;
    return p;
  };
  __bf16* xb    = (__bf16*)carve((size_t)MROWS * HID * 2);
  __bf16* wqkvb = (__bf16*)carve((size_t)QKV_O * HID * 2);
  __bf16* wob   = (__bf16*)carve((size_t)HID * HID * 2);
  float*  qkvf  = (float*) carve((size_t)MROWS * QKV_O * 4);
  __bf16* qb    = (__bf16*)carve((size_t)MROWS * NH * DH * 2);
  __bf16* kb    = (__bf16*)carve((size_t)MROWS * NKVH * DH * 2);
  __bf16* vtb   = (__bf16*)carve((size_t)MROWS * NKVH * DH * 2);
  __bf16* ctxb  = (__bf16*)carve((size_t)MROWS * HID * 2);

  const long nx = (long)MROWS * HID;
  const long nwqkv = (long)QKV_O * HID;
  const long nwo = (long)HID * HID;
  cvt_f32_bf16<<<(unsigned)((nx + 255) / 256), 256, 0, stream>>>(x, xb, nx);
  cvt_f32_bf16<<<(unsigned)((nwqkv + 255) / 256), 256, 0, stream>>>(Wqkv, wqkvb, nwqkv);
  cvt_f32_bf16<<<(unsigned)((nwo + 255) / 256), 256, 0, stream>>>(Wo, wob, nwo);

  // qkv = x @ Wqkv^T : (4096 x 8192), K = 6144
  gemm_bf16_nt<<<dim3(QKV_O / 512, MROWS / 32), 256, 0, stream>>>(
      xb, wqkvb, qkvf, MROWS, QKV_O, HID);

  // rope + split (2^25 elements)
  rope_split_kernel<<<(unsigned)(((long)MROWS * QKV_O) / 256), 256, 0, stream>>>(
      qkvf, qb, kb, vtb);

  // flash attention: 12288 waves, 4 waves per block
  attn_kernel<<<BATCH * NH * (SLEN / 16) / 4, 128, 0, stream>>>(qb, kb, vtb, ctxb);

  // out = ctx @ Wo^T : (4096 x 6144), K = 6144
  gemm_bf16_nt<<<dim3(HID / 512, MROWS / 32), 256, 0, stream>>>(
      ctxb, wob, out, MROWS, HID, HID);
}
